// MultiHeadAttention_40742059769886
// MI455X (gfx1250) — matrix operand order repaired and checked by reading, not compiled, not run on hardware
//
#include <hip/hip_runtime.h>

// Problem constants (reference: N=4096, D=256, H=4, DH=64, TOPK=30, SCALE=0.25)
#define N_TOK  4096
#define DMODEL 256
#define NHEAD  4
#define DHEAD  64
#define TOPK   30
#define SCALE_F 0.25f
#define LN_EPS 1e-5f

typedef __attribute__((ext_vector_type(16))) __bf16 v16bf;
typedef __attribute__((ext_vector_type(8)))  float  v8f;

union BF16x16 { unsigned int u[8]; v16bf v; };

__device__ inline unsigned short f2bf(float f) {
    unsigned int u = __float_as_uint(f);
    u += 0x7FFFu + ((u >> 16) & 1u);            // round-to-nearest-even
    return (unsigned short)(u >> 16);
}
__device__ inline unsigned int pack2(float lo, float hi) {
    return (unsigned int)f2bf(lo) | ((unsigned int)f2bf(hi) << 16);
}

// A-matrix fragment, 16x32 bf16, row-major source with leading dim `ld` (elems).
// lane m=L&15, hi=L>>4; VGPR v -> k=hi*16+2v (contiguous k order, matches load_b_pairs)
__device__ inline v16bf load_a_bf16(const unsigned short* tile, int ld, int lane) {
    const int m = lane & 15, hi = lane >> 4;
    const unsigned short* rp = tile + m * ld;
    BF16x16 r;
#pragma unroll
    for (int v = 0; v < 8; ++v) {
        const int k = hi * 16 + 2 * v;
        r.u[v] = *reinterpret_cast<const unsigned int*>(rp + k);
    }
    return r.v;
}

// B-matrix fragment, 32x16 bf16, from k-pair-packed dwords: pairs[kpair*ld + n]
// packs B[2*kpair][n] (lo) and B[2*kpair+1][n] (hi).
// lane n=L&15, hi=L>>4; VGPR v holds k-pair index hi*8+v (contiguous k order, this code's choice).
__device__ inline v16bf load_b_pairs(const unsigned int* pairs, int ld, int lane) {
    const int n = lane & 15, hi = lane >> 4;
    BF16x16 r;
#pragma unroll
    for (int v = 0; v < 8; ++v) r.u[v] = pairs[(hi * 8 + v) * ld + n];
    return r.v;
}

__device__ inline v8f wmma_bf16(v16bf a, v16bf b, v8f c) {
    return __builtin_amdgcn_wmma_f32_16x16x32_bf16(false, a, false, b, (short)0, c,
                                                   false, false);
}

// wave32 argmax reduce (ties -> lower index)
__device__ inline void red_max(float& v, int& i) {
#pragma unroll
    for (int off = 16; off > 0; off >>= 1) {
        float ov = __shfl_xor(v, off, 32);
        int   oi = __shfl_xor(i, off, 32);
        if (ov > v || (ov == v && oi < i)) { v = ov; i = oi; }
    }
}

// ---------------------------------------------- fp32 -> bf16 (pairs, coalesced)
__global__ void cvt_bf16_pairs_kernel(const float* __restrict__ in,
                                      unsigned int* __restrict__ out, int npairs) {
    int i = blockIdx.x * blockDim.x + threadIdx.x;
    if (i < npairs) out[i] = pack2(in[2 * i], in[2 * i + 1]);
}

// ---------------- weight W[d][c] (row-major) -> wp[c2][d] dword = (W[d][2c2],W[d][2c2+1])
__global__ void pack_weight_kernel(const float* __restrict__ W,
                                   unsigned int* __restrict__ wp) {
    int t = blockIdx.x * blockDim.x + threadIdx.x;    // t < 128*256
    int d = t & 255, c2 = t >> 8;
    const float* wr = W + d * DMODEL + 2 * c2;
    wp[c2 * DMODEL + d] = pack2(wr[0], wr[1]);
}

// ------------------------------------------------ projection GEMM (bf16 WMMA)
// out[n][d] = sum_c A[n][c] * W[d][c] + bias[d]   (A bf16, wp packed-transposed W)
__global__ __launch_bounds__(256) void proj_gemm_kernel(
    const unsigned short* __restrict__ Abf, const unsigned int* __restrict__ wp,
    const float* __restrict__ bias, float* __restrict__ out) {
    const int tid = threadIdx.x, w = tid >> 5, lane = tid & 31;
    const int hi = lane >> 4, col = lane & 15;
    const int n0 = blockIdx.x * 16;
    const int d0a = w * 16, d0b = (w + 8) * 16;
    v8f acc0 = {0.f, 0.f, 0.f, 0.f, 0.f, 0.f, 0.f, 0.f};
    v8f acc1 = acc0;
#pragma unroll
    for (int kc = 0; kc < 8; ++kc) {
        v16bf a  = load_a_bf16(Abf + n0 * DMODEL + kc * 32, DMODEL, lane);
        v16bf b0 = load_b_pairs(wp + (kc * 16) * DMODEL + d0a, DMODEL, lane);
        v16bf b1 = load_b_pairs(wp + (kc * 16) * DMODEL + d0b, DMODEL, lane);
        acc0 = wmma_bf16(a, b0, acc0);
        acc1 = wmma_bf16(a, b1, acc1);
    }
#pragma unroll
    for (int r = 0; r < 8; ++r) {
        const int n = n0 + r + 8 * hi;
        out[n * DMODEL + d0a + col] = acc0[r] + bias[d0a + col];
        out[n * DMODEL + d0b + col] = acc1[r] + bias[d0b + col];
    }
}

// ----- repack projections into attention layouts:
//  qh bf16 [h][n][dh] ; kp dword [h*32 + dh/2][m] (k-pair packed K^T) ; vh f32 [h][n][dh]
__global__ void repack_kernel(const float* __restrict__ qf, const float* __restrict__ kf,
                              const float* __restrict__ vf,
                              unsigned short* __restrict__ qh,
                              unsigned int* __restrict__ kp, float* __restrict__ vh) {
    const int t = blockIdx.x * blockDim.x + threadIdx.x;  // t < N*D
    const int n = t >> 8, dc = t & 255, h = dc >> 6, dh = dc & 63;
    const int hn = (h << 12) + n;
    qh[hn * DHEAD + dh] = f2bf(qf[t]);
    vh[hn * DHEAD + dh] = vf[t];
    if (!(dc & 1))
        kp[((h << 5) + (dh >> 1)) * N_TOK + n] = pack2(kf[t], kf[t + 1]);
}

// --------------------- fused attention: scores (WMMA, LDS-resident) + top-30
//   (cached local-max selection, cooperative rescan) + softmax + sparse PV
//   + dense attn row write-out (zeros + scatter) + ctx (bf16)
__global__ __launch_bounds__(512) void attn_topk_kernel(
    const unsigned short* __restrict__ qh, const unsigned int* __restrict__ kp,
    const float* __restrict__ vh, unsigned short* __restrict__ ctx,
    float* __restrict__ attn) {
    extern __shared__ float sm[];
    float* scores = sm;                     // 16 * 4096 f32  (256 KB)
    float* tv     = sm + 16 * N_TOK;        // 16 * 32 f32
    int*   ti     = (int*)(tv + 16 * 32);   // 16 * 32 i32

    const int tid = threadIdx.x, w = tid >> 5, lane = tid & 31;
    const int hi = lane >> 4, col = lane & 15;
    const int h = blockIdx.y, n0 = blockIdx.x * 16;

    // ---- Phase 1: S[16 x 4096] = (Q tile) x K^T, scaled, into LDS.
    const unsigned short* qbase = qh + (size_t)(h * N_TOK + n0) * DHEAD;
    const v16bf a0 = load_a_bf16(qbase, DHEAD, lane);        // dh 0..31
    const v16bf a1 = load_a_bf16(qbase + 32, DHEAD, lane);   // dh 32..63
    const unsigned int* kph = kp + (size_t)(h * 32) * N_TOK;
    for (int t = 0; t < 16; ++t) {
        const int m0 = w * 256 + t * 16;
        v16bf b0 = load_b_pairs(kph + m0, N_TOK, lane);
        v16bf b1 = load_b_pairs(kph + 16 * N_TOK + m0, N_TOK, lane);
        v8f acc = {0.f, 0.f, 0.f, 0.f, 0.f, 0.f, 0.f, 0.f};
        acc = wmma_bf16(a0, b0, acc);
        acc = wmma_bf16(a1, b1, acc);
#pragma unroll
        for (int r = 0; r < 8; ++r)
            scores[(r + 8 * hi) * N_TOK + m0 + col] = acc[r] * SCALE_F;
    }
    __syncthreads();

    // ---- Phase 2: one wave per query row.
    // Ownership map: owner(i) = ((i&63) + (i>>6)) & 31 — skewed so one lane's
    // 128 elements spread across LDS banks (cooperative rescan ~conflict-free).
    float* rowp = scores + w * N_TOK;
    float* tvr = tv + w * 32;
    int*   tir = ti + w * 32;
    const float NINF = -__builtin_inff();

    // initial per-lane local max over owned elements (conflict-free)
    float lm = NINF; int li = 0;
    for (int g = 0; g < 64; ++g) {
        const int i0 = 64 * g + ((lane - g) & 31);
        const float v0 = rowp[i0], v1 = rowp[i0 + 32];
        if (v0 > lm) { lm = v0; li = i0; }
        if (v1 > lm) { lm = v1; li = i0 + 32; }
    }
    float bm = lm; int bi = li;
    red_max(bm, bi);

    for (int t = 0; t < TOPK; ++t) {
        if (lane == 0) { tvr[t] = bm; tir[t] = bi; rowp[bi] = NINF; }
        const int wl = ((bi & 63) + (bi >> 6)) & 31;   // owner of extracted elem
        // cooperatively recompute lane wl's local max (4 loads/lane)
        float cm = NINF; int ci = 0;
#pragma unroll
        for (int m = 0; m < 2; ++m) {
            const int g = 2 * lane + m;
            const int i0 = 64 * g + ((wl - g) & 31);
            const float v0 = rowp[i0], v1 = rowp[i0 + 32];
            if (v0 > cm) { cm = v0; ci = i0; }
            if (v1 > cm) { cm = v1; ci = i0 + 32; }
        }
        red_max(cm, ci);
        if (lane == wl) { lm = cm; li = ci; }
        bm = lm; bi = li;
        red_max(bm, bi);
    }

    // softmax over the 30 kept scores
    const float maxv = tvr[0];
    float sum = 0.f;
    for (int j = 0; j < TOPK; ++j) {
        float e = __expf(tvr[j] - maxv);
        sum += e;
        if (lane == 0) tvr[j] = e;
    }
    const float inv = 1.f / sum;

    // sparse PV gather (fp32 V)
    float acc0 = 0.f, acc1 = 0.f;
    for (int j = 0; j < TOPK; ++j) {
        const int idx = tir[j];
        const float p = tvr[j] * inv;
        const float* vr = vh + (size_t)(h * N_TOK + idx) * DHEAD;
        acc0 += p * vr[lane];
        acc1 += p * vr[lane + 32];
    }

    // dense attn row: stream zeros (coalesced b128), then scatter 30 probs.
    // Same-wave store->store ordering guarantees the scatter lands last.
    const int n = n0 + w;
    float* arow = attn + ((size_t)h * N_TOK + n) * N_TOK;
    const float4 z = make_float4(0.f, 0.f, 0.f, 0.f);
#pragma unroll 4
    for (int m2 = 0; m2 < 32; ++m2)
        *reinterpret_cast<float4*>(arow + m2 * 128 + lane * 4) = z;
    if (lane < TOPK) arow[tir[lane]] = tvr[lane] * inv;

    unsigned short* cr = ctx + (size_t)n * DMODEL + h * DHEAD;
    cr[lane]      = f2bf(acc0);
    cr[lane + 32] = f2bf(acc1);
}

// -------------------- output projection (WMMA) + residual + LayerNorm (fused)
__global__ __launch_bounds__(256) void outproj_ln_kernel(
    const unsigned short* __restrict__ ctx, const unsigned int* __restrict__ wpo,
    const float* __restrict__ bo, const float* __restrict__ qin,
    const float* __restrict__ ln_g, const float* __restrict__ ln_b,
    float* __restrict__ out) {
    __shared__ float tile[16 * DMODEL];
    const int tid = threadIdx.x, w = tid >> 5, lane = tid & 31;
    const int hi = lane >> 4, col = lane & 15;
    const int n0 = blockIdx.x * 16;
    const int d0a = w * 16, d0b = (w + 8) * 16;
    v8f acc0 = {0.f, 0.f, 0.f, 0.f, 0.f, 0.f, 0.f, 0.f};
    v8f acc1 = acc0;
#pragma unroll
    for (int kc = 0; kc < 8; ++kc) {
        v16bf a  = load_a_bf16(ctx + n0 * DMODEL + kc * 32, DMODEL, lane);
        v16bf b0 = load_b_pairs(wpo + (kc * 16) * DMODEL + d0a, DMODEL, lane);
        v16bf b1 = load_b_pairs(wpo + (kc * 16) * DMODEL + d0b, DMODEL, lane);
        acc0 = wmma_bf16(a, b0, acc0);
        acc1 = wmma_bf16(a, b1, acc1);
    }
#pragma unroll
    for (int r = 0; r < 8; ++r) {
        const int m = r + 8 * hi, n = n0 + m;
        int d = d0a + col;
        tile[m * DMODEL + d] = acc0[r] + bo[d] + qin[n * DMODEL + d];
        d = d0b + col;
        tile[m * DMODEL + d] = acc1[r] + bo[d] + qin[n * DMODEL + d];
    }
    __syncthreads();
#pragma unroll
    for (int rr = 2 * w; rr < 2 * w + 2; ++rr) {
        const float* rp = tile + rr * DMODEL;
        float s1 = 0.f, s2 = 0.f;
        for (int i = lane; i < DMODEL; i += 32) {
            float x = rp[i];
            s1 += x;
            s2 += x * x;
        }
#pragma unroll
        for (int off = 16; off > 0; off >>= 1) {
            s1 += __shfl_xor(s1, off, 32);
            s2 += __shfl_xor(s2, off, 32);
        }
        const float mean = s1 * (1.f / DMODEL);
        const float var  = s2 * (1.f / DMODEL) - mean * mean;   // biased, torch-style
        const float rstd = rsqrtf(var + LN_EPS);
        const int n = n0 + rr;
        for (int i = lane; i < DMODEL; i += 32) {
            float x = rp[i];
            out[n * DMODEL + i] = (x - mean) * rstd * ln_g[i] + ln_b[i];
        }
    }
}

// ------------------------------------------------------------------- launcher
extern "C" void kernel_launch(void* const* d_in, const int* in_sizes, int n_in,
                              void* d_out, int out_size, void* d_ws, size_t ws_size,
                              hipStream_t stream) {
    (void)in_sizes; (void)n_in; (void)out_size; (void)ws_size;
    const float* key_in   = (const float*)d_in[0];
    const float* value_in = (const float*)d_in[1];
    const float* query_in = (const float*)d_in[2];
    const float* Wq = (const float*)d_in[3];  const float* bq = (const float*)d_in[4];
    const float* Wk = (const float*)d_in[5];  const float* bk = (const float*)d_in[6];
    const float* Wv = (const float*)d_in[7];  const float* bv = (const float*)d_in[8];
    const float* Wo = (const float*)d_in[9];  const float* bo = (const float*)d_in[10];
    const float* ln_g = (const float*)d_in[11];
    const float* ln_b = (const float*)d_in[12];

    float* out  = (float*)d_out;
    float* attn = out + (size_t)N_TOK * DMODEL;   // H*N*N fp32

    // workspace partition (256B aligned)
    char* ws = (char*)d_ws;
    size_t off = 0;
    auto alloc = [&](size_t bytes) -> void* {
        void* p = ws + off;
        off += (bytes + 255) & ~(size_t)255;
        return p;
    };
    const size_t ND = (size_t)N_TOK * DMODEL;
    unsigned short* qin_bf = (unsigned short*)alloc(ND * 2);
    unsigned short* kin_bf = (unsigned short*)alloc(ND * 2);
    unsigned short* vin_bf = (unsigned short*)alloc(ND * 2);
    unsigned int* wpq = (unsigned int*)alloc(128 * 256 * 4);
    unsigned int* wpk = (unsigned int*)alloc(128 * 256 * 4);
    unsigned int* wpv = (unsigned int*)alloc(128 * 256 * 4);
    unsigned int* wpo = (unsigned int*)alloc(128 * 256 * 4);
    float* qf = (float*)alloc(ND * 4);
    float* kf = (float*)alloc(ND * 4);
    float* vf = (float*)alloc(ND * 4);
    unsigned short* qh = (unsigned short*)alloc((size_t)NHEAD * N_TOK * DHEAD * 2);
    unsigned int*   kp = (unsigned int*)alloc((size_t)NHEAD * 32 * N_TOK * 4);
    float*          vh = (float*)alloc((size_t)NHEAD * N_TOK * DHEAD * 4);
    unsigned short* ctx = (unsigned short*)alloc(ND * 2);

    // 1) inputs -> bf16
    {
        const int npairs = (int)(ND / 2);
        cvt_bf16_pairs_kernel<<<npairs / 256, 256, 0, stream>>>(query_in, (unsigned int*)qin_bf, npairs);
        cvt_bf16_pairs_kernel<<<npairs / 256, 256, 0, stream>>>(key_in,   (unsigned int*)kin_bf, npairs);
        cvt_bf16_pairs_kernel<<<npairs / 256, 256, 0, stream>>>(value_in, (unsigned int*)vin_bf, npairs);
    }
    // 2) weights -> transposed, k-pair packed bf16
    pack_weight_kernel<<<128, 256, 0, stream>>>(Wq, wpq);
    pack_weight_kernel<<<128, 256, 0, stream>>>(Wk, wpk);
    pack_weight_kernel<<<128, 256, 0, stream>>>(Wv, wpv);
    pack_weight_kernel<<<128, 256, 0, stream>>>(Wo, wpo);
    // 3) Q/K/V projections (WMMA)
    proj_gemm_kernel<<<N_TOK / 16, 256, 0, stream>>>(qin_bf, wpq, bq, qf);
    proj_gemm_kernel<<<N_TOK / 16, 256, 0, stream>>>(kin_bf, wpk, bk, kf);
    proj_gemm_kernel<<<N_TOK / 16, 256, 0, stream>>>(vin_bf, wpv, bv, vf);
    // 4) repack into attention layouts
    repack_kernel<<<(int)(ND / 256), 256, 0, stream>>>(qf, kf, vf, qh, kp, vh);
    // 5) fused attention (scores WMMA + top-30 + softmax + sparse PV + attn rows)
    {
        dim3 grid(N_TOK / 16, NHEAD, 1);
        const size_t smem = (size_t)16 * N_TOK * 4 + 16 * 32 * 4 + 16 * 32 * 4; // 266240 B
        attn_topk_kernel<<<grid, 512, smem, stream>>>(qh, kp, vh, ctx, attn);
    }
    // 6) output projection + residual + LayerNorm
    outproj_ln_kernel<<<N_TOK / 16, 256, 0, stream>>>(ctx, wpo, bo, query_in, ln_g, ln_b, out);
}
